// ModalityTransformer_37967510897170
// MI455X (gfx1250) — matrix operand order repaired and checked by reading, not compiled, not run on hardware
//
#include <hip/hip_runtime.h>
#include <math.h>

// ------------------------------------------------------------------
// CDNA5 (gfx1250, wave32) implementation of the 2-layer modality
// transformer decoder.  All heavy math runs through
// v_wmma_f32_16x16x32_f16 (f16 A/B, f32 accumulate).
// ------------------------------------------------------------------

typedef _Float16 v16h __attribute__((ext_vector_type(16)));
typedef _Float16 h4   __attribute__((ext_vector_type(4)));
typedef float    v8f  __attribute__((ext_vector_type(8)));

static constexpr int Bc   = 32;
static constexpr int Tc   = 1024;
static constexpr int Dc   = 512;
static constexpr int DFFc = 2048;
static constexpr int Lc   = 2;
static constexpr int Hc   = 8;
static constexpr int Nc   = 4;
static constexpr int BSc  = Bc * Tc;     // 32768 independent "tokens"
static constexpr int Mc   = Nc * BSc;    // 131072 GEMM rows

// LDS tile stride in halves: 32 K-halves + 8 pad -> 80B row pitch,
// 16B aligned, no 4-way bank conflicts on b128 fragment reads.
static constexpr int LDSS = 40;

// ------------------------------------------------------------------
// C[M,Nout] = act( A[M,K] @ W[Nout,K]^T + bias[Nout] (+ Res[M,Nout]) )
// grid = (Nout/128, M/128), block = 256 threads = 8 waves.
// Wave w: rows [ (w&3)*32 , +32 ), cols [ (w>>2)*64 , +64 ).
// ------------------------------------------------------------------
template<bool RELU, bool RES>
__global__ __launch_bounds__(256)
void gemm_f16wmma(const float* __restrict__ A, const float* __restrict__ W,
                  const float* __restrict__ bias, const float* __restrict__ Res,
                  float* __restrict__ C, int K, int ldc)
{
    __shared__ __align__(16) _Float16 shA[128 * LDSS];
    __shared__ __align__(16) _Float16 shB[128 * LDSS];

    const int tid  = threadIdx.x;
    const int lane = tid & 31;
    const int wv   = tid >> 5;
    const int wrow = wv & 3;        // 4 row groups of 32
    const int wcol = wv >> 2;       // 2 col groups of 64
    const int l15  = lane & 15;
    const int hi   = lane >> 4;     // upper half-wave flag

    const size_t tileM = (size_t)blockIdx.y * 128;
    const size_t tileN = (size_t)blockIdx.x * 128;

    const float* Ab = A + tileM * (size_t)K;
    const float* Wb = W + tileN * (size_t)K;

    v8f acc[2][4];
    const v8f vzero = {0.f,0.f,0.f,0.f,0.f,0.f,0.f,0.f};
#pragma unroll
    for (int i = 0; i < 2; ++i)
#pragma unroll
        for (int j = 0; j < 4; ++j) acc[i][j] = vzero;

    // A fragment (16x32 f16): lanes 0-15 hold K {0..7,16..23}, lanes 16-31 {8..15,24..31}
    const int kb1 = hi ? 8 : 0;
    // B fragment (32x16 f16): lanes 0-15 hold K {0..7,16..23}, lanes 16-31 {8..15,24..31}
    const int kb2 = hi ? 8 : 0;

    for (int k0 = 0; k0 < K; k0 += 32) {
        // ---- stage 128x32 of A and of W (f32 -> f16) ----
        float4 ra[4], rb[4];
#pragma unroll
        for (int i = 0; i < 4; ++i) {
            const int c   = tid + 256 * i;      // 0..1023 chunk id
            const int row = c >> 3;
            const int col = (c & 7) << 2;
            ra[i] = *(const float4*)(Ab + (size_t)row * K + (k0 + col));
            rb[i] = *(const float4*)(Wb + (size_t)row * K + (k0 + col));
        }
        __syncthreads();                         // previous iter's reads done
#pragma unroll
        for (int i = 0; i < 4; ++i) {
            const int c   = tid + 256 * i;
            const int row = c >> 3;
            const int col = (c & 7) << 2;
            h4 ha = { (_Float16)ra[i].x, (_Float16)ra[i].y,
                      (_Float16)ra[i].z, (_Float16)ra[i].w };
            h4 hb = { (_Float16)rb[i].x, (_Float16)rb[i].y,
                      (_Float16)rb[i].z, (_Float16)rb[i].w };
            *(h4*)&shA[row * LDSS + col] = ha;
            *(h4*)&shB[row * LDSS + col] = hb;
        }
        __syncthreads();

        // ---- fragments from LDS (b128 reads) ----
        v16h af[2], bf[4];
#pragma unroll
        for (int i = 0; i < 2; ++i) {
            const _Float16* p = &shA[(wrow * 32 + i * 16 + l15) * LDSS + kb1];
            ((float4*)&af[i])[0] = *(const float4*)(p);        // K group 0
            ((float4*)&af[i])[1] = *(const float4*)(p + 16);   // K group +16
        }
#pragma unroll
        for (int j = 0; j < 4; ++j) {
            const _Float16* p = &shB[(wcol * 64 + j * 16 + l15) * LDSS + kb2];
            ((float4*)&bf[j])[0] = *(const float4*)(p);
            ((float4*)&bf[j])[1] = *(const float4*)(p + 16);
        }

        // ---- 8 WMMAs: 32x64 output per wave per K-step ----
#pragma unroll
        for (int i = 0; i < 2; ++i)
#pragma unroll
            for (int j = 0; j < 4; ++j)
                acc[i][j] = __builtin_amdgcn_wmma_f32_16x16x32_f16(
                    false, af[i], false, bf[j], (short)0, acc[i][j], false, false);
    }

    // ---- epilogue: bias (+residual) (+relu) ----
    // C/D layout: lane n=l15 (+0), vgpr r -> m = r + 8*hi
#pragma unroll
    for (int i = 0; i < 2; ++i)
#pragma unroll
        for (int j = 0; j < 4; ++j) {
            const size_t nn = tileN + (size_t)(wcol * 64 + j * 16 + l15);
            const float  bv = bias[nn];
#pragma unroll
            for (int r = 0; r < 8; ++r) {
                const size_t mm = tileM + (size_t)(wrow * 32 + i * 16 + r + hi * 8);
                float val = acc[i][j][r] + bv;
                if (RES)  val += Res[mm * (size_t)ldc + nn];
                if (RELU) val = fmaxf(val, 0.f);
                C[mm * (size_t)ldc + nn] = val;
            }
        }
}

// ------------------------------------------------------------------
// Tiny causal attention over N=4 modalities; one wave per (bs, head).
// QKV row layout: [q(512) | k(512) | v(512)], ldq = 1536.
// ------------------------------------------------------------------
__global__ __launch_bounds__(256)
void attn_kernel(const float* __restrict__ QKV, int ldq,
                 float* __restrict__ Out, int BS)
{
    const int lane = threadIdx.x & 31;
    const int wv   = threadIdx.x >> 5;
    const int idx  = blockIdx.x * 8 + wv;   // bs*H + h
    const int bs   = idx >> 3;              // H == 8
    const int h    = idx & 7;

    float2 q[4], k[4], v[4];
#pragma unroll
    for (int n = 0; n < 4; ++n) {
        const float* p = QKV + ((size_t)n * BS + bs) * (size_t)ldq + h * 64 + lane * 2;
        q[n] = *(const float2*)(p);
        k[n] = *(const float2*)(p + 512);
        v[n] = *(const float2*)(p + 1024);
    }

    float s[4][4];
#pragma unroll
    for (int i = 0; i < 4; ++i)
#pragma unroll
        for (int j = 0; j < 4; ++j) {
            if (j > i) continue;                       // causal mask
            float partial = q[i].x * k[j].x + q[i].y * k[j].y;
#pragma unroll
            for (int off = 16; off; off >>= 1)
                partial += __shfl_xor(partial, off, 32);
            s[i][j] = partial * 0.125f;                // 1/sqrt(64)
        }

#pragma unroll
    for (int i = 0; i < 4; ++i) {
        float mx = s[i][0];
        for (int j = 1; j <= i; ++j) mx = fmaxf(mx, s[i][j]);
        float pj[4]; float sum = 0.f;
        for (int j = 0; j <= i; ++j) { pj[j] = __expf(s[i][j] - mx); sum += pj[j]; }
        const float inv = 1.f / sum;
        float ox = 0.f, oy = 0.f;
        for (int j = 0; j <= i; ++j) { ox += pj[j] * v[j].x; oy += pj[j] * v[j].y; }
        float2 o; o.x = ox * inv; o.y = oy * inv;
        *(float2*)(Out + ((size_t)i * BS + bs) * 512 + h * 64 + lane * 2) = o;
    }
}

// ------------------------------------------------------------------
// LayerNorm over d=512; one wave per row (8 rows / block).
// ------------------------------------------------------------------
__global__ __launch_bounds__(256)
void ln_kernel(const float* __restrict__ Xin, const float* __restrict__ g,
               const float* __restrict__ b, float* __restrict__ Out)
{
    const int lane = threadIdx.x & 31;
    const int wv   = threadIdx.x >> 5;
    const size_t row = (size_t)blockIdx.x * 8 + wv;
    const float* x = Xin + row * 512;

    float4 vv[4];
    float sum = 0.f, ss = 0.f;
#pragma unroll
    for (int i = 0; i < 4; ++i) {
        vv[i] = *(const float4*)(x + i * 128 + lane * 4);
        sum += vv[i].x + vv[i].y + vv[i].z + vv[i].w;
        ss  += vv[i].x * vv[i].x + vv[i].y * vv[i].y
             + vv[i].z * vv[i].z + vv[i].w * vv[i].w;
    }
#pragma unroll
    for (int off = 16; off; off >>= 1) {
        sum += __shfl_xor(sum, off, 32);
        ss  += __shfl_xor(ss,  off, 32);
    }
    const float mu  = sum * (1.f / 512.f);
    const float var = ss  * (1.f / 512.f) - mu * mu;
    const float rs  = rsqrtf(var + 1e-5f);

    float* o = Out + row * 512;
#pragma unroll
    for (int i = 0; i < 4; ++i) {
        const int c = i * 128 + lane * 4;
        const float4 gg = *(const float4*)(g + c);
        const float4 bb = *(const float4*)(b + c);
        float4 r;
        r.x = (vv[i].x - mu) * rs * gg.x + bb.x;
        r.y = (vv[i].y - mu) * rs * gg.y + bb.y;
        r.z = (vv[i].z - mu) * rs * gg.z + bb.z;
        r.w = (vv[i].w - mu) * rs * gg.w + bb.w;
        *(float4*)(o + c) = r;
    }
}

// ------------------------------------------------------------------
// Stack 4 [BS*d] tensors into one [4, BS, d] buffer (float4 copies).
// ------------------------------------------------------------------
__global__ void pack4(const float* __restrict__ s0, const float* __restrict__ s1,
                      const float* __restrict__ s2, const float* __restrict__ s3,
                      float* __restrict__ dst, size_t n4)
{
    const size_t i = (size_t)blockIdx.x * blockDim.x + threadIdx.x;
    if (i >= n4) return;
    float4* D = (float4*)dst;
    D[i]          = ((const float4*)s0)[i];
    D[n4 + i]     = ((const float4*)s1)[i];
    D[2 * n4 + i] = ((const float4*)s2)[i];
    D[3 * n4 + i] = ((const float4*)s3)[i];
}

// ------------------------------------------------------------------
extern "C" void kernel_launch(void* const* d_in, const int* in_sizes, int n_in,
                              void* d_out, int out_size, void* d_ws, size_t ws_size,
                              hipStream_t stream)
{
    (void)in_sizes; (void)n_in; (void)out_size; (void)ws_size;

    const float* ctx0  = (const float*)d_in[0];
    const float* ctx1  = (const float*)d_in[1];
    const float* ctx2  = (const float*)d_in[2];
    const float* hctx  = (const float*)d_in[3];
    const float* tgt0  = (const float*)d_in[4];
    const float* tgt1  = (const float*)d_in[5];
    const float* tgt2  = (const float*)d_in[6];
    const float* tgt3  = (const float*)d_in[7];
    const float* Wqkv_s = (const float*)d_in[8];
    const float* bqkv_s = (const float*)d_in[9];
    const float* Wo_s   = (const float*)d_in[10];
    const float* bo_s   = (const float*)d_in[11];
    const float* Wqkv_c = (const float*)d_in[12];
    const float* bqkv_c = (const float*)d_in[13];
    const float* Wo_c   = (const float*)d_in[14];
    const float* bo_c   = (const float*)d_in[15];
    const float* W1     = (const float*)d_in[16];
    const float* b1     = (const float*)d_in[17];
    const float* W2     = (const float*)d_in[18];
    const float* b2     = (const float*)d_in[19];
    const float* lng    = (const float*)d_in[20];
    const float* lnb    = (const float*)d_in[21];

    // workspace carve-up (floats):
    //   X   [M,512]   current activation
    //   MEM [M,512]   packed cross-attn memory
    //   T1  [M,512]   attention output
    //   T2  [M,512]   pre-LN (proj + residual)
    //   FF  [M,2048]  QKV (ldc 1536) / FFN hidden (ldc 2048)
    float* X   = (float*)d_ws;
    float* MEM = X   + (size_t)Mc * Dc;
    float* T1  = MEM + (size_t)Mc * Dc;
    float* T2  = T1  + (size_t)Mc * Dc;
    float* FF  = T2  + (size_t)Mc * Dc;

    // pack modality stacks
    {
        const size_t n4 = (size_t)BSc * Dc / 4;
        const int blocks = (int)((n4 + 255) / 256);
        pack4<<<blocks, 256, 0, stream>>>(tgt0, tgt1, tgt2, tgt3, X, n4);
        pack4<<<blocks, 256, 0, stream>>>(hctx, ctx0, ctx1, ctx2, MEM, n4);
    }

    const dim3 blk(256);
    const int  gy = Mc / 128;                 // 1024 row-tiles
    const dim3 gAttn(BSc * Hc / 8);
    const dim3 gLN(Mc / 8);

    for (int l = 0; l < Lc; ++l) {
        const float* Wqs = Wqkv_s + (size_t)l * 3 * Dc * Dc;
        const float* bqs = bqkv_s + (size_t)l * 3 * Dc;
        const float* Wos = Wo_s   + (size_t)l * Dc * Dc;
        const float* bos = bo_s   + (size_t)l * Dc;
        const float* Wqc = Wqkv_c + (size_t)l * 3 * Dc * Dc;
        const float* bqc = bqkv_c + (size_t)l * 3 * Dc;
        const float* Woc = Wo_c   + (size_t)l * Dc * Dc;
        const float* boc = bo_c   + (size_t)l * Dc;
        const float* W1l = W1 + (size_t)l * DFFc * Dc;
        const float* b1l = b1 + (size_t)l * DFFc;
        const float* W2l = W2 + (size_t)l * Dc * DFFc;
        const float* b2l = b2 + (size_t)l * Dc;

        // ---- self-attention ----
        gemm_f16wmma<false,false><<<dim3(3*Dc/128, gy), blk, 0, stream>>>(
            X, Wqs, bqs, nullptr, FF, Dc, 3*Dc);
        attn_kernel<<<gAttn, blk, 0, stream>>>(FF, 3*Dc, T1, BSc);
        gemm_f16wmma<false,true><<<dim3(Dc/128, gy), blk, 0, stream>>>(
            T1, Wos, bos, X, T2, Dc, Dc);
        ln_kernel<<<gLN, blk, 0, stream>>>(
            T2, lng + (size_t)(l*3 + 0) * Dc, lnb + (size_t)(l*3 + 0) * Dc, X);

        // ---- cross-attention (Q from x, K/V from mem, same QKV buffer) ----
        gemm_f16wmma<false,false><<<dim3(Dc/128, gy), blk, 0, stream>>>(
            X, Wqc, bqc, nullptr, FF, Dc, 3*Dc);                       // cols 0..511
        gemm_f16wmma<false,false><<<dim3(2*Dc/128, gy), blk, 0, stream>>>(
            MEM, Wqc + (size_t)Dc * Dc, bqc + Dc, nullptr, FF + Dc, Dc, 3*Dc); // cols 512..1535
        attn_kernel<<<gAttn, blk, 0, stream>>>(FF, 3*Dc, T1, BSc);
        gemm_f16wmma<false,true><<<dim3(Dc/128, gy), blk, 0, stream>>>(
            T1, Woc, boc, X, T2, Dc, Dc);
        ln_kernel<<<gLN, blk, 0, stream>>>(
            T2, lng + (size_t)(l*3 + 1) * Dc, lnb + (size_t)(l*3 + 1) * Dc, X);

        // ---- FFN ----
        gemm_f16wmma<true,false><<<dim3(DFFc/128, gy), blk, 0, stream>>>(
            X, W1l, b1l, nullptr, FF, Dc, DFFc);
        gemm_f16wmma<false,true><<<dim3(Dc/128, gy), blk, 0, stream>>>(
            FF, W2l, b2l, X, T2, DFFc, Dc);
        float* lnout = (l == Lc - 1) ? (float*)d_out : X;
        ln_kernel<<<gLN, blk, 0, stream>>>(
            T2, lng + (size_t)(l*3 + 2) * Dc, lnb + (size_t)(l*3 + 2) * Dc, lnout);
    }
}
